// DDNN_37675453120668
// MI455X (gfx1250) — matrix operand order repaired and checked by reading, not compiled, not run on hardware
//
#include <hip/hip_runtime.h>
#include <hip/hip_bf16.h>

// ---------------- CDNA5 vector types ----------------
typedef float v8f  __attribute__((ext_vector_type(8)));
typedef __bf16 bf16_t;
typedef bf16_t v16bf __attribute__((ext_vector_type(16)));
typedef unsigned int v4u __attribute__((ext_vector_type(4)));
typedef int v4i __attribute__((ext_vector_type(4)));
typedef int v8i __attribute__((ext_vector_type(8)));

// ---------------- problem constants ----------------
constexpr int kB = 16;
constexpr int kC = 128;      // channels (M and K of the GEMM)
constexpr int kL = 32768;    // length (N of the GEMM)
constexpr int kG = 4;        // gate widths
constexpr int NCHUNK = 64;   // columns per LDS tile
constexpr int CHUNKS = 8;    // chunks per block (amortize A load)
constexpr int PITCH  = 136;  // bf16 elements per transposed LDS row (128 + 8 pad)

__device__ __forceinline__ unsigned short f2bf(float f) {
    unsigned int u = __float_as_uint(f);
    unsigned int r = u + 0x7FFFu + ((u >> 16) & 1u);   // round-to-nearest-even
    return (unsigned short)(r >> 16);
}
__device__ __forceinline__ float bf2f(unsigned short s) {
    return __uint_as_float(((unsigned int)s) << 16);
}

// ---------------------------------------------------------------------
// Tensor Data Mover: DMA a [kC rows x NCHUNK cols] fp32 tile (row stride
// kL elements) from global into LDS at byte offset lds_off.
// D# built per CDNA5 ISA 8.3/8.4: count=1, type=2, data_size=4B,
// tile_dim0=NCHUNK, tile_dim1=kC, tensor_dim0_stride=kL, 2D (groups 2/3 null).
// ---------------------------------------------------------------------
__device__ __forceinline__ void tdm_load_tile(unsigned lds_off, const float* gptr) {
    unsigned long long ga = (unsigned long long)(uintptr_t)gptr;
    v4u g0;
    g0.x = 1u;                                            // count=1 (valid user D#)
    g0.y = lds_off;                                       // lds_addr (bytes)
    g0.z = (unsigned)(ga & 0xFFFFFFFFu);                  // global_addr[31:0]
    g0.w = (unsigned)((ga >> 32) & 0x1FFFFFFu) | (2u << 30); // global_addr[56:32] | type=2
    v8i g1;
    g1[0] = (int)(2u << 16);                              // wg_mask=0, data_size=2 (4B)
    g1[1] = (int)(((unsigned)kL & 0xFFFFu) << 16);        // tensor_dim0[15:0] = 32768
    g1[2] = (int)(((unsigned)kC & 0xFFFFu) << 16);        // tensor_dim0[31:16]=0 | tensor_dim1[15:0]=128
    g1[3] = (int)((unsigned)NCHUNK << 16);                // tensor_dim1[31:16]=0 | tile_dim0=64
    g1[4] = (int)kC;                                      // tile_dim1=128 | tile_dim2=0
    g1[5] = (int)kL;                                      // tensor_dim0_stride[31:0]
    g1[6] = 0;                                            // stride[47:32]=0 | dim1_stride[15:0]=0
    g1[7] = 0;
    v4i z4 = {0, 0, 0, 0};
#if defined(__clang_major__) && (__clang_major__ >= 23)
    v8i z8 = {0, 0, 0, 0, 0, 0, 0, 0};
    __builtin_amdgcn_tensor_load_to_lds(g0, g1, z4, z4, z8, 0);
#else
    __builtin_amdgcn_tensor_load_to_lds(g0, g1, z4, z4, 0);
#endif
}

// =====================================================================
// Kernel 1: mean over L for each (b, c) row.  268 MB streamed read.
// =====================================================================
__global__ __launch_bounds__(256) void pool_kernel(const float* __restrict__ x,
                                                   float* __restrict__ pool) {
    __shared__ float red[256];
    const int rc = blockIdx.x;                       // b*kC + c  (2048 rows)
    const float4* r4 = (const float4*)(x + (size_t)rc * kL);
    float s = 0.f;
    for (int i = threadIdx.x; i < kL / 4; i += 256) {
        float4 v = r4[i];
        s += v.x + v.y + v.z + v.w;
    }
    red[threadIdx.x] = s;
    __syncthreads();
    for (int off = 128; off > 0; off >>= 1) {
        if (threadIdx.x < off) red[threadIdx.x] += red[threadIdx.x + off];
        __syncthreads();
    }
    if (threadIdx.x == 0) pool[rc] = red[0] * (1.0f / kL);
}

// =====================================================================
// Kernel 2: gate logits -> gumbel softmax (straight-through) ->
// per-(b,c) mult / mean / rstd.  Tiny (16 blocks x 128 threads).
// =====================================================================
__global__ __launch_bounds__(kC) void gate_kernel(const float* __restrict__ pool,
                                                  const float* __restrict__ gumbel,
                                                  const float* __restrict__ w_gate,
                                                  const float* __restrict__ rmean,
                                                  const float* __restrict__ rvar,
                                                  float* __restrict__ mult,
                                                  float* __restrict__ mean,
                                                  float* __restrict__ rstd) {
    __shared__ float part[kG][kC];
    __shared__ float gate_s[kG];
    const int b = blockIdx.x;
    const int c = threadIdx.x;
    const float p = pool[b * kC + c];
    for (int g = 0; g < kG; ++g) part[g][c] = p * w_gate[g * kC + c];
    __syncthreads();
    if (c < kG) {
        float s = 0.f;
        for (int i = 0; i < kC; ++i) s += part[c][i];
        part[c][0] = s;                               // logits[c]
    }
    __syncthreads();
    if (c == 0) {
        float y[kG];
        for (int g = 0; g < kG; ++g) y[g] = part[g][0] + gumbel[b * kG + g]; // TAU = 1
        float m = y[0];
        for (int g = 1; g < kG; ++g) m = fmaxf(m, y[g]);
        float e[kG], se = 0.f;
        for (int g = 0; g < kG; ++g) { e[g] = expf(y[g] - m); se += e[g]; }
        float soft[kG];
        for (int g = 0; g < kG; ++g) soft[g] = e[g] / se;
        int idx = 0; float best = soft[0];
        for (int g = 1; g < kG; ++g) if (soft[g] > best) { best = soft[g]; idx = g; }
        for (int g = 0; g < kG; ++g) {
            const float h = (g == idx) ? 1.f : 0.f;
            gate_s[g] = (h - soft[g]) + soft[g];      // straight-through, fp order kept
        }
    }
    __syncthreads();
    const int CH[kG] = {32, 64, 96, 128};
    float mu = 0.f, me = 0.f, va = 0.f;
    for (int g = 0; g < kG; ++g) {
        const float gv = gate_s[g];
        mu += gv * ((c < CH[g]) ? 1.f : 0.f);
        me += gv * rmean[g * kC + c];
        va += gv * rvar[g * kC + c];
    }
    mult[b * kC + c] = mu;
    mean[b * kC + c] = me;
    rstd[b * kC + c] = 1.0f / sqrtf(va + 1e-5f);
}

// =====================================================================
// Kernel 3: per-batch 128x128 @ 128xL GEMM via bf16x3 split-precision
// WMMA (v_wmma_f32_16x16x32_bf16).  x tiles are DMA'd into LDS by the
// Tensor Data Mover (double-buffered, TENSORcnt-tracked), then split
// into bf16 hi/lo transposed LDS for B-fragment ds_load_b128 reads.
// Block = 256 threads = 8 waves; wave w owns output rows 16w..16w+15.
// =====================================================================
__global__ __launch_bounds__(256) void gemm_kernel(const float* __restrict__ x,
                                                   const float* __restrict__ conv_w,
                                                   const float* __restrict__ conv_b,
                                                   const float* __restrict__ bn_w,
                                                   const float* __restrict__ bn_b,
                                                   const float* __restrict__ mult_w,
                                                   const float* __restrict__ mean_w,
                                                   const float* __restrict__ rstd_w,
                                                   float* __restrict__ out) {
    __shared__ float stage[2][kC * NCHUNK];          // raw fp32 tiles, TDM dest (2x32KB)
    __shared__ unsigned short hiT[NCHUNK * PITCH];   // x hi, transposed [n][k]
    __shared__ unsigned short loT[NCHUNK * PITCH];   // x lo, transposed [n][k]

    const int b       = blockIdx.y;
    const int n_base  = blockIdx.x * (NCHUNK * CHUNKS);
    const int tid     = threadIdx.x;
    const int wave    = tid >> 5;
    const int lane    = tid & 31;
    const int lane16  = lane & 15;
    const int hi_half = lane >> 4;                   // 0: lanes 0-15, 1: lanes 16-31
    const int m0      = wave * 16;

    const unsigned stage_off[2] = { (unsigned)(uintptr_t)&stage[0][0],
                                    (unsigned)(uintptr_t)&stage[1][0] };

    // ---- persistent A fragments: conv_w rows m0..m0+15, split bf16 hi/lo ----
    // v16bf A k order used here (same as B): lanes0-15 hold K {k0..k0+15},
    // lanes16-31 hold K {k0+16..k0+31}.
    v16bf a_hi[4], a_lo[4];
    {
        const int row   = m0 + lane16;
        const int kbase = hi_half * 16;
        for (int s = 0; s < 4; ++s) {
            union { v16bf v; unsigned short u[16]; } Ah, Al;
            const float* wr = conv_w + row * kC + s * 32 + kbase;
            #pragma unroll
            for (int j = 0; j < 8; ++j) {
                const float f0 = wr[j];
                const float f1 = wr[8 + j];
                const unsigned short h0 = f2bf(f0);
                const unsigned short h1 = f2bf(f1);
                Ah.u[j]     = h0;
                Ah.u[8 + j] = h1;
                Al.u[j]     = f2bf(f0 - bf2f(h0));
                Al.u[8 + j] = f2bf(f1 - bf2f(h1));
            }
            a_hi[s] = Ah.v;
            a_lo[s] = Al.v;
        }
    }

    const float* xb = x   + (size_t)b * kC * kL;
    float*       ob = out + (size_t)b * kC * kL;
    const int    bC = b * kC;

    // ---- prologue: DMA chunk 0 into stage[0] (one TDM issue per block) ----
    if (wave == 0) tdm_load_tile(stage_off[0], xb + n_base);

    for (int ch = 0; ch < CHUNKS; ++ch) {
        const int l0  = n_base + ch * NCHUNK;
        const int buf = ch & 1;

        // ---- prefetch chunk ch+1 via TDM, then wait for chunk ch ----
        if (wave == 0) {
            if (ch + 1 < CHUNKS) {
                tdm_load_tile(stage_off[(ch + 1) & 1], xb + l0 + NCHUNK);
                __builtin_amdgcn_s_wait_tensorcnt(1);   // chunk ch landed (in-order)
            } else {
                __builtin_amdgcn_s_wait_tensorcnt(0);   // last chunk: drain all
            }
        }
        __syncthreads();                             // stage[buf] ready; hi/lo free

        // ---- split stage[buf] fp32 -> bf16 hi/lo, transposed LDS ----
        {
            const int n  = tid & 63;
            const int kb = tid >> 6;                 // 0..3
            const float* st = &stage[buf][0];
            #pragma unroll 4
            for (int p = 0; p < 32; ++p) {
                const int k = p * 4 + kb;
                const float f = st[k * NCHUNK + n];
                const unsigned short h = f2bf(f);
                hiT[n * PITCH + k] = h;
                loT[n * PITCH + k] = f2bf(f - bf2f(h));
            }
        }
        __syncthreads();

        v8f acc[4];
        #pragma unroll
        for (int nt = 0; nt < 4; ++nt) acc[nt] = (v8f){0.f,0.f,0.f,0.f,0.f,0.f,0.f,0.f};

        // ---- K loop: 4 steps of 32, 3 WMMAs per tile (hi*hi + hi*lo + lo*hi) ----
        #pragma unroll
        for (int s = 0; s < 4; ++s) {
            const int koff = s * 32 + hi_half * 16;  // B layout: lanes0-15 K0-15, lanes16-31 K16-31
            #pragma unroll
            for (int nt = 0; nt < 4; ++nt) {
                const int n = nt * 16 + lane16;
                union { v16bf v; v4u q[2]; } Bh, Bl;
                const v4u* ph = (const v4u*)&hiT[n * PITCH + koff];
                const v4u* pl = (const v4u*)&loT[n * PITCH + koff];
                Bh.q[0] = ph[0]; Bh.q[1] = ph[1];
                Bl.q[0] = pl[0]; Bl.q[1] = pl[1];
                acc[nt] = __builtin_amdgcn_wmma_f32_16x16x32_bf16(
                    false, a_hi[s], false, Bh.v, (short)0, acc[nt], false, false);
                acc[nt] = __builtin_amdgcn_wmma_f32_16x16x32_bf16(
                    false, a_hi[s], false, Bl.v, (short)0, acc[nt], false, false);
                acc[nt] = __builtin_amdgcn_wmma_f32_16x16x32_bf16(
                    false, a_lo[s], false, Bh.v, (short)0, acc[nt], false, false);
            }
        }

        // ---- fused epilogue: +bias, NaN-mask, gate, BN, NaN-mask, gate ----
        // C/D layout: VGPR r -> (M = r + 8*hi_half, N = lane16)
        #pragma unroll
        for (int r = 0; r < 8; ++r) {
            const int row = m0 + r + hi_half * 8;
            const float mu = mult_w[bC + row];
            const float me = mean_w[bC + row];
            const float rs = rstd_w[bC + row];
            const float bw = bn_w[row];
            const float bb = bn_b[row];
            const float cb = conv_b[row];
            float* orow = ob + (size_t)row * kL + l0 + lane16;
            #pragma unroll
            for (int nt = 0; nt < 4; ++nt) {
                float v = acc[nt][r] + cb;
                v = (v != v) ? 0.0f : v;             // NaN mask (pre-gate)
                v *= mu;                             // gate
                float xn = (v - me) * rs * bw + bb;  // dynamic BN
                xn = (xn != xn) ? 0.0f : xn;         // NaN mask
                __builtin_nontemporal_store(xn * mu, orow + nt * 16);
            }
        }
    }
}

// =====================================================================
extern "C" void kernel_launch(void* const* d_in, const int* in_sizes, int n_in,
                              void* d_out, int out_size, void* d_ws, size_t ws_size,
                              hipStream_t stream) {
    (void)in_sizes; (void)n_in; (void)out_size; (void)ws_size;
    const float* x      = (const float*)d_in[0];
    const float* gumbel = (const float*)d_in[1];
    const float* w_gate = (const float*)d_in[2];
    const float* conv_w = (const float*)d_in[3];
    const float* conv_b = (const float*)d_in[4];
    const float* bn_w   = (const float*)d_in[5];
    const float* bn_b   = (const float*)d_in[6];
    const float* rmean  = (const float*)d_in[7];
    const float* rvar   = (const float*)d_in[8];
    float* out = (float*)d_out;

    float* W    = (float*)d_ws;          // workspace: 4 * 2048 floats = 32 KB
    float* pool = W;
    float* mult = W + kB * kC;
    float* mean = W + 2 * kB * kC;
    float* rstd = W + 3 * kB * kC;

    pool_kernel<<<kB * kC, 256, 0, stream>>>(x, pool);
    gate_kernel<<<kB, kC, 0, stream>>>(pool, gumbel, w_gate, rmean, rvar, mult, mean, rstd);
    dim3 grid(kL / (NCHUNK * CHUNKS), kB);           // 64 x 16 = 1024 blocks
    gemm_kernel<<<grid, 256, 0, stream>>>(x, conv_w, conv_b, bn_w, bn_b,
                                          mult, mean, rstd, out);
}
